// LatentSpaceMLA_17051020165273
// MI455X (gfx1250) — matrix operand order repaired and checked by reading, not compiled, not run on hardware
//
#include <hip/hip_runtime.h>
#include <hip/hip_bf16.h>
#include <math.h>

// ---------------------------------------------------------------------------
// MLA prefill for gfx1250 (MI455X): bf16 WMMA everywhere, f32 accumulate.
// Attention tiles are staged into LDS by the Tensor Data Mover (TDM).
// ---------------------------------------------------------------------------

typedef __bf16 bf16_t;
typedef bf16_t bf16x8  __attribute__((ext_vector_type(8)));
typedef bf16_t bf16x16 __attribute__((ext_vector_type(16)));
typedef float  floatx8 __attribute__((ext_vector_type(8)));
typedef unsigned int u32x4 __attribute__((ext_vector_type(4)));
typedef int i32x4 __attribute__((ext_vector_type(4)));
typedef int i32x8 __attribute__((ext_vector_type(8)));

#define SEQ      1024
#define D_MODEL  7168
#define NUM_HEADS 128
#define Q_LORA   1536
#define KV_LORA  512
#define DQ       192
#define DV       128
#define QK_COLS  (NUM_HEADS * DQ)   /* 24576 */
#define V_COLS   (NUM_HEADS * DV)   /* 16384 */

static __device__ inline unsigned short f2bf(float f) {
  unsigned u = __float_as_uint(f);
  u += 0x7FFFu + ((u >> 16) & 1u);      // round-to-nearest-even
  return (unsigned short)(u >> 16);
}
static __device__ inline float bf2f(unsigned short h) {
  return __uint_as_float(((unsigned)h) << 16);
}

// 16-bit A fragment (16x32, MxK): lanes 0-15 = rows m, v0..3 hold K 0..7 /
// hi-half K 8..15, v4..7 hold K 16..23 / 24..31 (per CDNA5 ISA 7.12.2).
static __device__ inline bf16x16 frag_a(const unsigned short* base, long ld, int lane) {
  const int m  = lane & 15;
  const int hi = lane >> 4;
  const unsigned short* p = base + (long)m * ld + hi * 8;
  bf16x8 lo = *(const bf16x8*)(p);
  bf16x8 hh = *(const bf16x8*)(p + 16);
  return __builtin_shufflevector(lo, hh, 0,1,2,3,4,5,6,7,8,9,10,11,12,13,14,15);
}
// B fragment from a [N,K] row-major operand (i.e. Bmath = operand^T, KxN):
// lane n = column n; v0..3 hold K 0..7 / hi-half K 8..15, v4..7 hold K 16..23 / 24..31.
static __device__ inline bf16x16 frag_b(const unsigned short* base, long ld, int lane) {
  const int n  = lane & 15;
  const int hi = lane >> 4;
  const unsigned short* p = base + (long)n * ld + hi * 8;
  bf16x8 lo = *(const bf16x8*)(p);
  bf16x8 hh = *(const bf16x8*)(p + 16);
  return __builtin_shufflevector(lo, hh, 0,1,2,3,4,5,6,7,8,9,10,11,12,13,14,15);
}

static __device__ inline floatx8 wmma_bf16(bf16x16 a, bf16x16 b, floatx8 c) {
  return __builtin_amdgcn_wmma_f32_16x16x32_bf16(false, a, false, b, (short)0, c,
                                                 false, false);
}

// ---------------------------------------------------------------------------
// TDM: DMA a 2D tile of 16-bit elements from global memory into LDS.
// D# per CDNA5 ISA ch.8: group0 {count=1, lds_addr, global_addr, type=2},
// group1 {mask=0, data_size=1(2B), tensor/tile dims + stride}, groups2/3 = 0.
// This toolchain declares the 6-arg builtin (extra int32x8 descriptor words).
// ---------------------------------------------------------------------------
static __device__ inline void tdm_load_2d_b16(unsigned lds_byte_off,
                                              const void* gptr,
                                              unsigned tensor_d0, unsigned tensor_d1,
                                              unsigned long long stride0_elems,
                                              unsigned tile_d0, unsigned tile_d1) {
  const unsigned long long ga = (unsigned long long)(uintptr_t)gptr;
  u32x4 g0;
  g0.x = 1u;                                            // count=1, user D#
  g0.y = lds_byte_off;                                  // lds_addr
  g0.z = (unsigned)(ga & 0xFFFFFFFFu);                  // global_addr[31:0]
  g0.w = (unsigned)((ga >> 32) & 0x01FFFFFFu)           // global_addr[56:32]
       | (2u << 30);                                    // type=2 ("image")
  i32x8 g1;
  g1[0] = (int)(1u << 16);                              // data_size=1 -> 2 bytes
  g1[1] = (int)((tensor_d0 & 0xFFFFu) << 16);           // tensor_dim0[15:0]
  g1[2] = (int)((tensor_d0 >> 16) | ((tensor_d1 & 0xFFFFu) << 16));
  g1[3] = (int)((tensor_d1 >> 16) | (tile_d0 << 16));   // tile_dim0
  g1[4] = (int)tile_d1;                                 // tile_dim1, tile_dim2=0
  g1[5] = (int)(stride0_elems & 0xFFFFFFFFu);           // tensor_dim0_stride lo
  g1[6] = (int)((stride0_elems >> 32) & 0xFFFFu);       // stride hi, stride1=0
  g1[7] = 0;
  const i32x4 z4 = (i32x4)0;
  const i32x8 z8 = (i32x8)0;
  __builtin_amdgcn_tensor_load_to_lds(g0, g1, z4, z4, z8, 0);
}

// ---------------------------------------------------------------------------
// fp32 -> bf16 streaming conversion
// ---------------------------------------------------------------------------
__global__ __launch_bounds__(256) void cvt_f32_bf16_kernel(
    const float* __restrict__ in, unsigned short* __restrict__ out, long n) {
  const long stride = (long)gridDim.x * blockDim.x;
  for (long i = (long)blockIdx.x * blockDim.x + threadIdx.x; i < n; i += stride)
    out[i] = f2bf(in[i]);
}

// ---------------------------------------------------------------------------
// Tiled bf16 GEMM: C = A[M,K] * B[N,K]^T, 128x128 block, K-step 32, 8 waves.
// MODE 0: bf16 C row-major; MODE 1: bf16 C^T ([N,M]); MODE 2: f32 C row-major.
// ---------------------------------------------------------------------------
#define BM 128
#define BN 128
#define BK 32
#define LDP 40   /* padded LDS row stride (elems) -> conflict-free b128 reads */

template <int MODE>
__global__ __launch_bounds__(256) void gemm_bf16_kernel(
    const unsigned short* __restrict__ A,
    const unsigned short* __restrict__ B,
    void* __restrict__ C,
    int M, int N, int K)
{
  __shared__ alignas(16) unsigned short sA[BM * LDP];
  __shared__ alignas(16) unsigned short sB[BN * LDP];
  const int tid  = threadIdx.x;
  const int lane = tid & 31;
  const int wv   = tid >> 5;
  const int wm   = wv >> 2;            // 0..1 : 64 rows of the block tile
  const int wn   = wv & 3;             // 0..3 : 32 cols of the block tile
  const long bm  = (long)blockIdx.y * BM;
  const long bn  = (long)blockIdx.x * BN;

  floatx8 acc[4][2];
#pragma unroll
  for (int i = 0; i < 4; ++i)
#pragma unroll
    for (int j = 0; j < 2; ++j) acc[i][j] = (floatx8)0.0f;

  const int crow = tid >> 1;           // 0..127
  const int ccol = (tid & 1) * 16;     // 0 or 16
  const unsigned short* ga = A + (bm + crow) * (long)K + ccol;
  const unsigned short* gb = B + (bn + crow) * (long)K + ccol;

  for (int k0 = 0; k0 < K; k0 += BK) {
    uint4 a0 = *(const uint4*)(ga + k0);
    uint4 a1 = *(const uint4*)(ga + k0 + 8);
    uint4 b0 = *(const uint4*)(gb + k0);
    uint4 b1 = *(const uint4*)(gb + k0 + 8);
    // WGP-scope prefetch of the next K tile (global_prefetch_b8)
    __builtin_prefetch(ga + k0 + BK, 0, 3);
    __builtin_prefetch(gb + k0 + BK, 0, 3);
    *(uint4*)&sA[crow * LDP + ccol]     = a0;
    *(uint4*)&sA[crow * LDP + ccol + 8] = a1;
    *(uint4*)&sB[crow * LDP + ccol]     = b0;
    *(uint4*)&sB[crow * LDP + ccol + 8] = b1;
    __syncthreads();

    bf16x16 af[4], bfr[2];
#pragma unroll
    for (int i = 0; i < 4; ++i)
      af[i] = frag_a(&sA[(wm * 64 + i * 16) * LDP], LDP, lane);
#pragma unroll
    for (int j = 0; j < 2; ++j)
      bfr[j] = frag_b(&sB[(wn * 32 + j * 16) * LDP], LDP, lane);
#pragma unroll
    for (int i = 0; i < 4; ++i)
#pragma unroll
      for (int j = 0; j < 2; ++j)
        acc[i][j] = wmma_bf16(af[i], bfr[j], acc[i][j]);
    __syncthreads();
  }

  const int nl = lane & 15;
  const int hi = lane >> 4;
#pragma unroll
  for (int i = 0; i < 4; ++i) {
#pragma unroll
    for (int j = 0; j < 2; ++j) {
      const long mb = bm + wm * 64 + i * 16 + hi * 8;   // C row for v=0
      const long nc = bn + wn * 32 + j * 16 + nl;       // C column
#pragma unroll
      for (int v = 0; v < 8; ++v) {
        const float x = acc[i][j][v];
        if (MODE == 0)      ((unsigned short*)C)[(mb + v) * N + nc] = f2bf(x);
        else if (MODE == 1) ((unsigned short*)C)[nc * (long)M + (mb + v)] = f2bf(x);
        else                ((float*)C)[(mb + v) * N + nc] = x;
      }
    }
  }
}

// ---------------------------------------------------------------------------
// Interleaved RoPE on the last 64 dims of each 192-wide head (q or k buffer).
// Pair i uses freq index (2i) mod 32 (matches reference emb[::2]).
// ---------------------------------------------------------------------------
__global__ __launch_bounds__(256) void rope_kernel(unsigned short* __restrict__ t) {
  const int idx = blockIdx.x * 256 + threadIdx.x;      // SEQ*128*32 threads
  const int i = idx & 31;
  const int h = (idx >> 5) & (NUM_HEADS - 1);
  const int s = idx >> 12;
  const int fi = (2 * i) & 31;
  const float invf = __powf(10000.0f, -(float)fi / 32.0f);
  const float th = (float)s * invf;
  float sn, c;
  __sincosf(th, &sn, &c);
  const long base = (long)s * QK_COLS + h * DQ + 128 + 2 * i;
  const float x1 = bf2f(t[base]);
  const float x2 = bf2f(t[base + 1]);
  t[base]     = f2bf(x1 * c - x2 * sn);
  t[base + 1] = f2bf(x1 * sn + x2 * c);
}

// ---------------------------------------------------------------------------
// Flash-style attention: 1 wave = 1 head x 16 query rows, 32 keys/step.
// K and V^T tiles are DMA'd into LDS by the TDM (TENSORcnt-tracked), then
// 12 WMMAs produce scores (K=192), online softmax runs with cross-lane shfl
// reductions, P is restaged through LDS, and 8 WMMAs accumulate O += P*V.
// ---------------------------------------------------------------------------
__global__ __launch_bounds__(32) void mla_attn_kernel(
    const unsigned short* __restrict__ q,
    const unsigned short* __restrict__ k,
    const unsigned short* __restrict__ vT,   // [NUM_HEADS*DV, SEQ]
    unsigned short* __restrict__ out)        // [SEQ, NUM_HEADS*DV]
{
  __shared__ alignas(16) unsigned short sK[32 * DQ];   // 32 keys x 192 dims
  __shared__ alignas(16) unsigned short sV[DV * 32];   // 128 dims x 32 keys
  __shared__ alignas(16) unsigned short sP[16 * 32];
  const int lane = threadIdx.x & 31;
  const int h  = blockIdx.x;
  const int qb = blockIdx.y * 16;
  const int nl = lane & 15;
  const int hi = lane >> 4;
  const float scale = 0.07216878364870322f;  // 1/sqrt(192)
  const unsigned sK_off = (unsigned)(uintptr_t)(void*)&sK[0];
  const unsigned sV_off = (unsigned)(uintptr_t)(void*)&sV[0];

  bf16x16 aq[6];
#pragma unroll
  for (int c = 0; c < 6; ++c)
    aq[c] = frag_a(q + (long)qb * QK_COLS + h * DQ + c * 32, QK_COLS, lane);

  floatx8 o[8];
  float mrun[8], lrun[8];
#pragma unroll
  for (int j = 0; j < 8; ++j) o[j] = (floatx8)0.0f;
#pragma unroll
  for (int v = 0; v < 8; ++v) { mrun[v] = -3.0e38f; lrun[v] = 0.0f; }

  for (int kb = 0; kb < SEQ; kb += 32) {
    // DMA K tile: 32 key rows x 192 dims, row stride QK_COLS.
    tdm_load_2d_b16(sK_off, k + (long)kb * QK_COLS + h * DQ,
                    DQ, 32, QK_COLS, DQ, 32);
    // DMA V^T tile: 128 dim rows x 32 keys, row stride SEQ.
    tdm_load_2d_b16(sV_off, vT + (long)h * DV * SEQ + kb,
                    32, DV, SEQ, 32, DV);
    __builtin_amdgcn_s_wait_tensorcnt((short)0);
    __syncthreads();

    floatx8 s0 = (floatx8)0.0f, s1 = (floatx8)0.0f;
#pragma unroll
    for (int c = 0; c < 6; ++c) {
      bf16x16 b0 = frag_b(sK + c * 32, DQ, lane);
      bf16x16 b1 = frag_b(sK + 16 * DQ + c * 32, DQ, lane);
      s0 = wmma_bf16(aq[c], b0, s0);
      s1 = wmma_bf16(aq[c], b1, s1);
    }
    float p0[8], p1[8];
#pragma unroll
    for (int v = 0; v < 8; ++v) {          // row (query) = v + 8*hi
      float a = s0[v] * scale;
      float b = s1[v] * scale;
      float mx = fmaxf(a, b);
      mx = fmaxf(mx, __shfl_xor(mx, 1, 32));
      mx = fmaxf(mx, __shfl_xor(mx, 2, 32));
      mx = fmaxf(mx, __shfl_xor(mx, 4, 32));
      mx = fmaxf(mx, __shfl_xor(mx, 8, 32));
      float mnew = fmaxf(mrun[v], mx);
      float corr = __expf(mrun[v] - mnew);
      float e0 = __expf(a - mnew);
      float e1 = __expf(b - mnew);
      float rs = e0 + e1;
      rs += __shfl_xor(rs, 1, 32);
      rs += __shfl_xor(rs, 2, 32);
      rs += __shfl_xor(rs, 4, 32);
      rs += __shfl_xor(rs, 8, 32);
      lrun[v] = lrun[v] * corr + rs;
      mrun[v] = mnew;
      p0[v] = e0; p1[v] = e1;
#pragma unroll
      for (int j = 0; j < 8; ++j) o[j][v] *= corr;
    }
#pragma unroll
    for (int v = 0; v < 8; ++v) {
      const int row = v + hi * 8;
      sP[row * 32 + nl]      = f2bf(p0[v]);
      sP[row * 32 + nl + 16] = f2bf(p1[v]);
    }
    __syncthreads();
    bf16x16 ap = frag_a(sP, 32, lane);
#pragma unroll
    for (int j = 0; j < 8; ++j) {
      bf16x16 bv = frag_b(sV + (j * 16) * 32, 32, lane);
      o[j] = wmma_bf16(ap, bv, o[j]);
    }
    __syncthreads();
  }
  float rl[8];
#pragma unroll
  for (int v = 0; v < 8; ++v) rl[v] = __frcp_rn(lrun[v]);
#pragma unroll
  for (int j = 0; j < 8; ++j) {
#pragma unroll
    for (int v = 0; v < 8; ++v) {
      const long row = qb + v + hi * 8;
      const long col = (long)h * DV + j * 16 + nl;
      out[row * V_COLS + col] = f2bf(o[j][v] * rl[v]);
    }
  }
}

// ---------------------------------------------------------------------------
// Orchestration
// ---------------------------------------------------------------------------
extern "C" void kernel_launch(void* const* d_in, const int* in_sizes, int n_in,
                              void* d_out, int out_size, void* d_ws, size_t ws_size,
                              hipStream_t stream) {
  (void)in_sizes; (void)n_in; (void)out_size; (void)ws_size;
  const float* hs   = (const float*)d_in[0];
  const float* Wqa  = (const float*)d_in[1];
  const float* Wkva = (const float*)d_in[2];
  const float* Wqb  = (const float*)d_in[3];
  const float* Wkb  = (const float*)d_in[4];
  const float* Wvb  = (const float*)d_in[5];
  const float* Wo   = (const float*)d_in[6];

  char* base = (char*)d_ws;
  size_t off = 0;
  auto take = [&](size_t elems) {
    unsigned short* p = (unsigned short*)(base + off);
    off = (off + elems * 2 + 255) & ~(size_t)255;
    return p;
  };
  unsigned short* hs_b   = take((size_t)SEQ * D_MODEL);
  unsigned short* wqa_b  = take((size_t)Q_LORA * D_MODEL);
  unsigned short* wkva_b = take((size_t)KV_LORA * D_MODEL);
  unsigned short* wqb_b  = take((size_t)QK_COLS * Q_LORA);
  unsigned short* wkb_b  = take((size_t)QK_COLS * KV_LORA);
  unsigned short* wvb_b  = take((size_t)V_COLS * KV_LORA);
  unsigned short* wo_b   = take((size_t)D_MODEL * V_COLS);
  unsigned short* qlat   = take((size_t)SEQ * Q_LORA);
  unsigned short* kvlat  = take((size_t)SEQ * KV_LORA);
  unsigned short* qbuf   = take((size_t)SEQ * QK_COLS);
  unsigned short* kbuf   = take((size_t)SEQ * QK_COLS);
  unsigned short* vTbuf  = take((size_t)V_COLS * SEQ);
  unsigned short* aobuf  = take((size_t)SEQ * V_COLS);

  auto cvt = [&](const float* in, unsigned short* outp, long n) {
    long blocks = (n + 255) / 256;
    if (blocks > 32768) blocks = 32768;
    cvt_f32_bf16_kernel<<<dim3((unsigned)blocks), dim3(256), 0, stream>>>(in, outp, n);
  };
  cvt(hs,   hs_b,   (long)SEQ * D_MODEL);
  cvt(Wqa,  wqa_b,  (long)Q_LORA * D_MODEL);
  cvt(Wkva, wkva_b, (long)KV_LORA * D_MODEL);
  cvt(Wqb,  wqb_b,  (long)QK_COLS * Q_LORA);
  cvt(Wkb,  wkb_b,  (long)QK_COLS * KV_LORA);
  cvt(Wvb,  wvb_b,  (long)V_COLS * KV_LORA);
  cvt(Wo,   wo_b,   (long)D_MODEL * V_COLS);

  gemm_bf16_kernel<0><<<dim3(Q_LORA / 128, SEQ / 128), dim3(256), 0, stream>>>(
      hs_b, wqa_b, qlat, SEQ, Q_LORA, D_MODEL);            // q latent
  gemm_bf16_kernel<0><<<dim3(KV_LORA / 128, SEQ / 128), dim3(256), 0, stream>>>(
      hs_b, wkva_b, kvlat, SEQ, KV_LORA, D_MODEL);         // kv latent
  gemm_bf16_kernel<0><<<dim3(QK_COLS / 128, SEQ / 128), dim3(256), 0, stream>>>(
      qlat, wqb_b, qbuf, SEQ, QK_COLS, Q_LORA);            // q up-proj
  gemm_bf16_kernel<0><<<dim3(QK_COLS / 128, SEQ / 128), dim3(256), 0, stream>>>(
      kvlat, wkb_b, kbuf, SEQ, QK_COLS, KV_LORA);          // k up-proj
  gemm_bf16_kernel<1><<<dim3(V_COLS / 128, SEQ / 128), dim3(256), 0, stream>>>(
      kvlat, wvb_b, vTbuf, SEQ, V_COLS, KV_LORA);          // v up-proj -> V^T

  rope_kernel<<<dim3(16384), dim3(256), 0, stream>>>(qbuf);
  rope_kernel<<<dim3(16384), dim3(256), 0, stream>>>(kbuf);

  mla_attn_kernel<<<dim3(NUM_HEADS, SEQ / 16), dim3(32), 0, stream>>>(
      qbuf, kbuf, vTbuf, aobuf);

  gemm_bf16_kernel<2><<<dim3(D_MODEL / 128, SEQ / 128), dim3(256), 0, stream>>>(
      aobuf, wo_b, d_out, SEQ, D_MODEL, V_COLS);           // output proj -> f32
}
